// PGN_49563922596335
// MI455X (gfx1250) — matrix operand order repaired and checked by reading, not compiled, not run on hardware
//
#include <hip/hip_runtime.h>
#include <hip/hip_bf16.h>

#define N_NODES 50000
#define N_EDGES 800000
#define DF 64
#define HF 128

typedef __attribute__((ext_vector_type(16))) _Float16 v16h;
typedef __attribute__((ext_vector_type(8)))  float    v8f;
typedef __attribute__((ext_vector_type(2)))  _Float16 h2;

// ---------------------------------------------------------------------------
// WMMA fragment loaders (wave32 layouts per CDNA5 ISA §7.12.2)

// A: 16x32 f16 from row-major tile (LDS), row stride lda halves (16B multiple).
// lane L: row m=L&15, K-base kb=(L>>4)*8; elems 0..7 -> K=kb+0..7,
// elems 8..15 -> K=kb+16..23.  Two aligned 16-byte chunks -> 2x ds_load_b128.
__device__ __forceinline__ v16h frag_A(const _Float16* base, int lda, int lane) {
    const int m  = lane & 15;
    const int kb = (lane >> 4) << 3;
    const _Float16* row = base + m * lda + kb;
    union { v16h v; uint4 q[2]; } r;
    r.q[0] = *(const uint4*)(row);
    r.q[1] = *(const uint4*)(row + 16);
    return r.v;
}

// B: pre-swizzled fragment-major weights: out[(frag*32+lane)*16 + e]
// -> two global_load_b128 per lane.
__device__ __forceinline__ v16h frag_B(const _Float16* wp, int frag, int lane) {
    const uint4* p = (const uint4*)(wp + ((size_t)(frag * 32 + lane) << 4));
    union { v16h v; uint4 q[2]; } r;
    r.q[0] = p[0];
    r.q[1] = p[1];
    return r.v;
}

__device__ __forceinline__ v8f wmma_f16(v16h a, v16h b, v8f c) {
    return __builtin_amdgcn_wmma_f32_16x16x32_f16(false, a, false, b,
                                                  (short)0, c, false, false);
}

__device__ __forceinline__ h2 cvt2(float2 a) {
    h2 t = { (_Float16)a.x, (_Float16)a.y };
    return t;
}

// ---------------------------------------------------------------------------
// Repack row-major f32 weights [K][N] into f16 WMMA B-fragment order:
//   frag = kt*(N/16)+nt ; out[(frag*32+lane)*16 + e] = W[kt*32+(lane>>4)*8+(e&7)+(e>>3)*16][nt*16+(lane&15)]
__global__ void GN_pack_B(const float* __restrict__ W, _Float16* __restrict__ out,
                          int K, int N) {
    const int i = blockIdx.x * blockDim.x + threadIdx.x;
    if (i >= K * N) return;
    const int e      = i & 15;
    const int lane   = (i >> 4) & 31;
    const int frag   = i >> 9;
    const int ntiles = N >> 4;
    const int nt = frag % ntiles;
    const int kt = frag / ntiles;
    const int k = kt * 32 + ((lane >> 4) << 3) + (e & 7) + ((e >> 3) << 4);
    const int n = nt * 16 + (lane & 15);
    out[i] = (_Float16)W[k * N + n];
}

__global__ void GN_zero_f32(float* __restrict__ p, int n) {
    const int i = blockIdx.x * blockDim.x + threadIdx.x;
    if (i < n) p[i] = 0.0f;
}

// ---------------------------------------------------------------------------
// Edge kernel: one wave per 16-edge tile (50000 tiles exactly).
//   updated_ef = MLP([nf[src], nf[dst], ef, u])   (256 -> 128 -> 64)
//   out_ef = updated_ef + ef ; agg[dst] += updated_ef ; deg[dst] += 1
//   edge_sum += column sums of updated_ef
__global__ __launch_bounds__(32) void GN_edge_kernel(
    const float* __restrict__ nf, const float* __restrict__ ef,
    const float* __restrict__ u,
    const _Float16* __restrict__ eW1p, const float* __restrict__ eb1,
    const _Float16* __restrict__ eW2p, const float* __restrict__ eb2,
    const int* __restrict__ src, const int* __restrict__ dst,
    float* __restrict__ out_ef, float* __restrict__ agg,
    float* __restrict__ deg, float* __restrict__ edge_sum)
{
    __shared__ __align__(16) _Float16 s_in[16][4 * DF]; // 8KB
    __shared__ __align__(16) _Float16 s_h [16][HF];     // 4KB
    __shared__ __align__(16) float    s_out[16][DF];    // 4KB
    __shared__ int s_srcv[16];
    __shared__ int s_dstv[16];

    const int lane = threadIdx.x;
    const int e0   = blockIdx.x * 16;

    if (lane < 16) {
        s_srcv[lane] = src[e0 + lane];
        const int d = dst[e0 + lane];
        s_dstv[lane] = d;
        atomicAdd(&deg[d], 1.0f);
    }
    __syncthreads();

    // ---- stage concat(nf[src], nf[dst], ef, u) as f16: 4x (float2 -> h2) ---
    const h2 uh = cvt2(((const float2*)u)[lane]);
#pragma unroll 1
    for (int e = 0; e < 16; ++e) {
        h2* row = (h2*)&s_in[e][0];
        const float2 a0 = ((const float2*)(nf + (size_t)s_srcv[e] * DF))[lane];
        const float2 a1 = ((const float2*)(nf + (size_t)s_dstv[e] * DF))[lane];
        const float2 a2 = ((const float2*)(ef + (size_t)(e0 + e) * DF))[lane];
        row[lane]          = cvt2(a0);
        row[32 + lane]     = cvt2(a1);
        row[64 + lane]     = cvt2(a2);
        row[96 + lane]     = uh;
    }
    __syncthreads();

    const int n  = lane & 15;
    const int mb = (lane >> 4) << 3;

    // ---- layer 1: [16x256]@[256x128], ReLU -> s_h --------------------------
#pragma unroll 1
    for (int nt = 0; nt < 8; ++nt) {
        v8f acc = {};
#pragma unroll
        for (int kt = 0; kt < 8; ++kt) {
            v16h a = frag_A(&s_in[0][0] + kt * 32, 4 * DF, lane);
            v16h b = frag_B(eW1p, kt * 8 + nt, lane);
            acc = wmma_f16(a, b, acc);
        }
        const float bias = eb1[nt * 16 + n];
#pragma unroll
        for (int i = 0; i < 8; ++i) {
            float v = acc[i] + bias;
            v = v > 0.0f ? v : 0.0f;
            s_h[mb + i][nt * 16 + n] = (_Float16)v;
        }
    }
    __syncthreads();

    // ---- layer 2: [16x128]@[128x64] + bias -> s_out, column sums -----------
#pragma unroll 1
    for (int nt = 0; nt < 4; ++nt) {
        v8f acc = {};
#pragma unroll
        for (int kt = 0; kt < 4; ++kt) {
            v16h a = frag_A(&s_h[0][0] + kt * 32, HF, lane);
            v16h b = frag_B(eW2p, kt * 4 + nt, lane);
            acc = wmma_f16(a, b, acc);
        }
        const int c = nt * 16 + n;
        const float bias = eb2[c];
        float colsum = 0.0f;
#pragma unroll
        for (int i = 0; i < 8; ++i) {
            const float v = acc[i] + bias;
            s_out[mb + i][c] = v;
            colsum += v;
        }
        colsum += __shfl_xor(colsum, 16, 32);
        if (lane < 16) atomicAdd(&edge_sum[c], colsum);
    }
    __syncthreads();

    // ---- writeback: residual (float4) + scatter-add into agg ---------------
#pragma unroll 1
    for (int t = lane; t < 16 * (DF / 4); t += 32) {
        const int e = t >> 4;          // edge row 0..15
        const int q = t & 15;          // float4 chunk 0..15
        const int edge = e0 + e;
        const float4 v  = ((const float4*)&s_out[e][0])[q];
        const float4 fe = ((const float4*)(ef + (size_t)edge * DF))[q];
        float4 o; o.x = v.x + fe.x; o.y = v.y + fe.y; o.z = v.z + fe.z; o.w = v.w + fe.w;
        ((float4*)(out_ef + (size_t)edge * DF))[q] = o;
        float* ag = agg + (size_t)s_dstv[e] * DF + q * 4;
        atomicAdd(ag + 0, v.x);
        atomicAdd(ag + 1, v.y);
        atomicAdd(ag + 2, v.z);
        atomicAdd(ag + 3, v.w);
    }
}

// ---------------------------------------------------------------------------
// Node kernel: one wave per 16-node tile (3125 tiles exactly).
//   updated_nf = MLP([agg/deg, nf, u])   (192 -> 128 -> 64)
__global__ __launch_bounds__(32) void GN_node_kernel(
    const float* __restrict__ nf, const float* __restrict__ u,
    const _Float16* __restrict__ nW1p, const float* __restrict__ nb1,
    const _Float16* __restrict__ nW2p, const float* __restrict__ nb2,
    const float* __restrict__ agg, const float* __restrict__ deg,
    float* __restrict__ out_nf, float* __restrict__ node_sum)
{
    __shared__ __align__(16) _Float16 s_in[16][3 * DF]; // 6KB
    __shared__ __align__(16) _Float16 s_h [16][HF];     // 4KB
    __shared__ __align__(16) float    s_out[16][DF];    // 4KB
    __shared__ float s_rdg[16];

    const int lane = threadIdx.x;
    const int n0   = blockIdx.x * 16;

    if (lane < 16) {
        float dg = deg[n0 + lane];
        dg = dg > 1.0f ? dg : 1.0f;
        s_rdg[lane] = 1.0f / dg;
    }
    __syncthreads();

    const h2 uh = cvt2(((const float2*)u)[lane]);
#pragma unroll 1
    for (int r = 0; r < 16; ++r) {
        const int node = n0 + r;
        const float rdg = s_rdg[r];
        h2* row = (h2*)&s_in[r][0];
        float2 a0 = ((const float2*)(agg + (size_t)node * DF))[lane];
        a0.x *= rdg; a0.y *= rdg;
        const float2 a1 = ((const float2*)(nf + (size_t)node * DF))[lane];
        row[lane]      = cvt2(a0);
        row[32 + lane] = cvt2(a1);
        row[64 + lane] = uh;
    }
    __syncthreads();

    const int n  = lane & 15;
    const int mb = (lane >> 4) << 3;

    // layer 1: [16x192]@[192x128]
#pragma unroll 1
    for (int nt = 0; nt < 8; ++nt) {
        v8f acc = {};
#pragma unroll
        for (int kt = 0; kt < 6; ++kt) {
            v16h a = frag_A(&s_in[0][0] + kt * 32, 3 * DF, lane);
            v16h b = frag_B(nW1p, kt * 8 + nt, lane);
            acc = wmma_f16(a, b, acc);
        }
        const float bias = nb1[nt * 16 + n];
#pragma unroll
        for (int i = 0; i < 8; ++i) {
            float v = acc[i] + bias;
            v = v > 0.0f ? v : 0.0f;
            s_h[mb + i][nt * 16 + n] = (_Float16)v;
        }
    }
    __syncthreads();

    // layer 2: [16x128]@[128x64]
#pragma unroll 1
    for (int nt = 0; nt < 4; ++nt) {
        v8f acc = {};
#pragma unroll
        for (int kt = 0; kt < 4; ++kt) {
            v16h a = frag_A(&s_h[0][0] + kt * 32, HF, lane);
            v16h b = frag_B(nW2p, kt * 4 + nt, lane);
            acc = wmma_f16(a, b, acc);
        }
        const int c = nt * 16 + n;
        const float bias = nb2[c];
        float colsum = 0.0f;
#pragma unroll
        for (int i = 0; i < 8; ++i) {
            const float v = acc[i] + bias;
            s_out[mb + i][c] = v;
            colsum += v;
        }
        colsum += __shfl_xor(colsum, 16, 32);
        if (lane < 16) atomicAdd(&node_sum[c], colsum);
    }
    __syncthreads();

    // writeback: residual with float4
#pragma unroll 1
    for (int t = lane; t < 16 * (DF / 4); t += 32) {
        const int r = t >> 4;
        const int q = t & 15;
        const int node = n0 + r;
        const float4 v  = ((const float4*)&s_out[r][0])[q];
        const float4 fn = ((const float4*)(nf + (size_t)node * DF))[q];
        float4 o; o.x = v.x + fn.x; o.y = v.y + fn.y; o.z = v.z + fn.z; o.w = v.w + fn.w;
        ((float4*)(out_nf + (size_t)node * DF))[q] = o;
    }
}

// ---------------------------------------------------------------------------
// Global kernel: one block. gm_in = [node_sum/N, edge_sum/E, u] -> MLP -> +u
__global__ __launch_bounds__(192) void GN_global_kernel(
    const float* __restrict__ u,
    const float* __restrict__ gW1, const float* __restrict__ gb1,
    const float* __restrict__ gW2, const float* __restrict__ gb2,
    const float* __restrict__ node_sum, const float* __restrict__ edge_sum,
    float* __restrict__ out_u)
{
    __shared__ float gin[3 * DF];
    __shared__ float hid[HF];
    const int t = threadIdx.x;
    if      (t < DF)      gin[t] = node_sum[t] * (1.0f / N_NODES);
    else if (t < 2 * DF)  gin[t] = edge_sum[t - DF] * (1.0f / N_EDGES);
    else                  gin[t] = u[t - 2 * DF];
    __syncthreads();
    if (t < HF) {
        float a = gb1[t];
#pragma unroll 4
        for (int k = 0; k < 3 * DF; ++k) a += gin[k] * gW1[k * HF + t];
        hid[t] = a > 0.0f ? a : 0.0f;
    }
    __syncthreads();
    if (t < DF) {
        float a = gb2[t];
#pragma unroll 4
        for (int j = 0; j < HF; ++j) a += hid[j] * gW2[j * DF + t];
        out_u[t] = a + u[t];
    }
}

// ---------------------------------------------------------------------------
extern "C" void kernel_launch(void* const* d_in, const int* in_sizes, int n_in,
                              void* d_out, int out_size, void* d_ws, size_t ws_size,
                              hipStream_t stream) {
    const float* nf  = (const float*)d_in[0];
    const float* ef  = (const float*)d_in[1];
    const float* u   = (const float*)d_in[2];
    const float* eW1 = (const float*)d_in[3];
    const float* eb1 = (const float*)d_in[4];
    const float* eW2 = (const float*)d_in[5];
    const float* eb2 = (const float*)d_in[6];
    const float* nW1 = (const float*)d_in[7];
    const float* nb1 = (const float*)d_in[8];
    const float* nW2 = (const float*)d_in[9];
    const float* nb2 = (const float*)d_in[10];
    const float* gW1 = (const float*)d_in[11];
    const float* gb1 = (const float*)d_in[12];
    const float* gW2 = (const float*)d_in[13];
    const float* gb2 = (const float*)d_in[14];
    const int*   src = (const int*)d_in[15];
    const int*   dst = (const int*)d_in[16];

    // d_out layout: [nf 50000x64][ef 800000x64][u 1x64], flat f32
    float* out_nf = (float*)d_out;
    float* out_ef = out_nf + (size_t)N_NODES * DF;
    float* out_u  = out_ef + (size_t)N_EDGES * DF;

    // workspace layout (bytes, 32B-aligned)
    char* ws = (char*)d_ws;
    _Float16* eW1p = (_Float16*)(ws + 0);         // 4D*H = 32768 f16
    _Float16* eW2p = (_Float16*)(ws + 65536);     // H*D  =  8192 f16
    _Float16* nW1p = (_Float16*)(ws + 81920);     // 3D*H = 24576 f16
    _Float16* nW2p = (_Float16*)(ws + 131072);    // H*D  =  8192 f16
    float* agg      = (float*)(ws + 147456);      // N*D f32 = 12.8 MB
    float* deg      = agg + (size_t)N_NODES * DF; // N f32
    float* edge_sum = deg + N_NODES;              // 64 f32
    float* node_sum = edge_sum + DF;              // 64 f32

    // 1) weight repack f32 -> f16 WMMA fragment order
    GN_pack_B<<<(4 * DF * HF + 255) / 256, 256, 0, stream>>>(eW1, eW1p, 4 * DF, HF);
    GN_pack_B<<<(HF * DF + 255) / 256, 256, 0, stream>>>(eW2, eW2p, HF, DF);
    GN_pack_B<<<(3 * DF * HF + 255) / 256, 256, 0, stream>>>(nW1, nW1p, 3 * DF, HF);
    GN_pack_B<<<(HF * DF + 255) / 256, 256, 0, stream>>>(nW2, nW2p, HF, DF);

    // 2) zero accumulators (agg, deg, edge_sum, node_sum contiguous)
    const int nzero = N_NODES * DF + N_NODES + 2 * DF;
    GN_zero_f32<<<(nzero + 255) / 256, 256, 0, stream>>>(agg, nzero);

    // 3) edge MLP + scatter aggregation
    GN_edge_kernel<<<N_EDGES / 16, 32, 0, stream>>>(
        nf, ef, u, eW1p, eb1, eW2p, eb2, src, dst,
        out_ef, agg, deg, edge_sum);

    // 4) node MLP + readout
    GN_node_kernel<<<N_NODES / 16, 32, 0, stream>>>(
        nf, u, nW1p, nb1, nW2p, nb2, agg, deg, out_nf, node_sum);

    // 5) global MLP
    GN_global_kernel<<<1, 192, 0, stream>>>(
        u, gW1, gb1, gW2, gb2, node_sum, edge_sum, out_u);
}
